// GraphSelfAttentionLayer_71330816852534
// MI455X (gfx1250) — matrix operand order repaired and checked by reading, not compiled, not run on hardware
//
#include <hip/hip_runtime.h>

typedef __attribute__((ext_vector_type(16))) __bf16 v16bf;
typedef __attribute__((ext_vector_type(8)))  float  v8f;

__device__ __forceinline__ v8f wmma_bf16(v16bf a, v16bf b, v8f c) {
  // 8 args: (neg_a, A, neg_b, B, c_mod, C, reuse_a, reuse_b)
  return __builtin_amdgcn_wmma_f32_16x16x32_bf16(
      false, a, false, b, (short)0, c, false, false);
}

// round-to-nearest-even float -> bf16 bits (explicit, no native-cast reliance)
__device__ __forceinline__ unsigned short f2bf_bits(float x) {
  unsigned u = __builtin_bit_cast(unsigned, x);
  return (unsigned short)((u + 0x7fffu + ((u >> 16) & 1u)) >> 16);
}
__device__ __forceinline__ float bfbits2f(unsigned short s) {
  return __builtin_bit_cast(float, ((unsigned)s) << 16);
}

#define GM 4608   // B*N rows
#define GK 1024   // D
#define GN 1024   // D (output cols)
#define KP 32     // K panel (one bf16 WMMA depth)
#define LDP 40    // padded LDS row pitch (bf16 elements); 80B, 16B-aligned
#define LPN (128 * 36 * 16 * 36)   // lp element count

// ---------------------------------------------------------------------------
// One-time fp32 -> (bf16 hi, bf16 lo) split. x = hi + lo to ~2^-16 relative.
// ---------------------------------------------------------------------------
__global__ __launch_bounds__(256) void split_kernel(
    const float* __restrict__ src, __bf16* __restrict__ dh,
    __bf16* __restrict__ dl, int n4)
{
  int i = blockIdx.x * 256 + threadIdx.x;
  if (i >= n4) return;
  float4 v = ((const float4*)src)[i];
  float x[4] = {v.x, v.y, v.z, v.w};
  unsigned short hb[4], lb[4];
  #pragma unroll
  for (int j = 0; j < 4; j++) {
    hb[j] = f2bf_bits(x[j]);
    lb[j] = f2bf_bits(x[j] - bfbits2f(hb[j]));
  }
  uint2 hw, lw;
  hw.x = (unsigned)hb[0] | ((unsigned)hb[1] << 16);
  hw.y = (unsigned)hb[2] | ((unsigned)hb[3] << 16);
  lw.x = (unsigned)lb[0] | ((unsigned)lb[1] << 16);
  lw.y = (unsigned)lb[2] | ((unsigned)lb[3] << 16);
  *(uint2*)(dh + (size_t)i * 4) = hw;
  *(uint2*)(dl + (size_t)i * 4) = lw;
}

// ---------------------------------------------------------------------------
// Kernel 1: P[z] = roi @ W[z]^T (+ bias[z]);  z=0:q 1:k 2:vproj
// fp32-faithful GEMM on the bf16 pipe: A*B ~= Ah*Bh + Ah*Bl + Al*Bh (f32 acc).
// 128 threads (4 waves), 64x64 tile, wave = 32x32 (2x2 WMMA frags).
// Register-buffered, LDS double-buffered K pipeline; staging is pure b128 moves.
// Staging uses individually named uint4 locals (no arrays/lambdas) so the
// values stay in VGPRs — no scratch spills.
// ---------------------------------------------------------------------------

#define LOAD_REGS(K0)                                              \
  do {                                                             \
    size_t o0 = (size_t)ch0_r * GK + (size_t)(K0) + ch0_c;         \
    size_t o1 = (size_t)ch1_r * GK + (size_t)(K0) + ch0_c;         \
    rAh0 = *(const uint4*)(pAh + o0);                              \
    rAh1 = *(const uint4*)(pAh + o1);                              \
    rAl0 = *(const uint4*)(pAl + o0);                              \
    rAl1 = *(const uint4*)(pAl + o1);                              \
    rBh0 = *(const uint4*)(pBh + o0);                              \
    rBh1 = *(const uint4*)(pBh + o1);                              \
    rBl0 = *(const uint4*)(pBl + o0);                              \
    rBl1 = *(const uint4*)(pBl + o1);                              \
  } while (0)

#define STORE_REGS(BUF)                                            \
  do {                                                             \
    *(uint4*)&sm[BUF][0][ch0_r][ch0_c] = rAh0;                     \
    *(uint4*)&sm[BUF][0][ch1_r][ch0_c] = rAh1;                     \
    *(uint4*)&sm[BUF][1][ch0_r][ch0_c] = rAl0;                     \
    *(uint4*)&sm[BUF][1][ch1_r][ch0_c] = rAl1;                     \
    *(uint4*)&sm[BUF][2][ch0_r][ch0_c] = rBh0;                     \
    *(uint4*)&sm[BUF][2][ch1_r][ch0_c] = rBh1;                     \
    *(uint4*)&sm[BUF][3][ch0_r][ch0_c] = rBl0;                     \
    *(uint4*)&sm[BUF][3][ch1_r][ch0_c] = rBl1;                     \
  } while (0)

__global__ __launch_bounds__(128) void proj_gemm_kernel(
    const __bf16* __restrict__ Ah_g, const __bf16* __restrict__ Al_g,
    const __bf16* __restrict__ Wh_g, const __bf16* __restrict__ Wl_g,
    const float*  __restrict__ bq,   const float*  __restrict__ bk,
    float* __restrict__ P)
{
  const int z = blockIdx.z;
  const size_t zoff = (size_t)z * GN * GK;
  const float* bias = (z == 0) ? bq : (z == 1) ? bk : nullptr;
  const int mbase = blockIdx.y * 64;
  const int nbase = blockIdx.x * 64;

  // planes: 0=Ah 1=Al 2=Bh 3=Bl
  __shared__ __align__(16) __bf16 sm[2][4][64][LDP];

  const int t      = threadIdx.x;
  const int lane   = t & 31;
  const int wv     = t >> 5;
  const int hi     = lane >> 4;
  const int l16    = lane & 15;
  const int rowsel = (wv >> 1) * 32;
  const int colsel = (wv & 1) * 32;

  // staging mapping: 256 16B-chunks per plane, 2 per thread
  const int ch0_r = (t + 0)   >> 2;          // 0..31
  const int ch1_r = (t + 128) >> 2;          // 32..63
  const int ch0_c = (t & 3) * 8;             // bf16 offset within row

  const __bf16* pAh = Ah_g + (size_t)mbase * GK;
  const __bf16* pAl = Al_g + (size_t)mbase * GK;
  const __bf16* pBh = Wh_g + zoff + (size_t)nbase * GK;
  const __bf16* pBl = Wl_g + zoff + (size_t)nbase * GK;

  uint4 rAh0, rAh1, rAl0, rAl1, rBh0, rBh1, rBl0, rBl1;

  v8f acc0 = v8f{0.f,0.f,0.f,0.f,0.f,0.f,0.f,0.f};
  v8f acc1 = acc0, acc2 = acc0, acc3 = acc0;

  // prologue: stage panel 0
  LOAD_REGS(0);
  STORE_REGS(0);
  __syncthreads();

  for (int k0 = 0; k0 < GK; k0 += KP) {
    const int buf = (k0 / KP) & 1;
    const bool has_next = (k0 + KP) < GK;
    if (has_next) {
      LOAD_REGS(k0 + KP);                      // global -> regs (overlaps compute)
      if (k0 + 2 * KP < GK) {                  // L2 prefetch for panel after next
        __builtin_prefetch(pAh + (size_t)ch0_r * GK + k0 + 2 * KP + ch0_c, 0, 1);
        __builtin_prefetch(pBh + (size_t)ch0_r * GK + k0 + 2 * KP + ch0_c, 0, 1);
      }
    }

    // Fragments. A 16x32 bf16: halves 0..7 -> K=8*hi+j ; 8..15 -> K=16+8*hi+j.
    v16bf ah0, ah1, al0, al1;
    #pragma unroll
    for (int j = 0; j < 8; j++) {
      ah0[j]     = sm[buf][0][rowsel      + l16][ 8*hi + j];
      ah0[8 + j] = sm[buf][0][rowsel      + l16][16 + 8*hi + j];
      ah1[j]     = sm[buf][0][rowsel + 16 + l16][ 8*hi + j];
      ah1[8 + j] = sm[buf][0][rowsel + 16 + l16][16 + 8*hi + j];
      al0[j]     = sm[buf][1][rowsel      + l16][ 8*hi + j];
      al0[8 + j] = sm[buf][1][rowsel      + l16][16 + 8*hi + j];
      al1[j]     = sm[buf][1][rowsel + 16 + l16][ 8*hi + j];
      al1[8 + j] = sm[buf][1][rowsel + 16 + l16][16 + 8*hi + j];
    }
    // B 32x16 bf16: halves 0..7 -> K=8*hi+j ; 8..15 -> K=16+8*hi+j, N = lane&15.
    v16bf bh0, bh1, bl0, bl1;
    #pragma unroll
    for (int j = 0; j < 8; j++) {
      bh0[j]     = sm[buf][2][colsel      + l16][ 8*hi + j];
      bh0[8 + j] = sm[buf][2][colsel      + l16][16 + 8*hi + j];
      bh1[j]     = sm[buf][2][colsel + 16 + l16][ 8*hi + j];
      bh1[8 + j] = sm[buf][2][colsel + 16 + l16][16 + 8*hi + j];
      bl0[j]     = sm[buf][3][colsel      + l16][ 8*hi + j];
      bl0[8 + j] = sm[buf][3][colsel      + l16][16 + 8*hi + j];
      bl1[j]     = sm[buf][3][colsel + 16 + l16][ 8*hi + j];
      bl1[8 + j] = sm[buf][3][colsel + 16 + l16][16 + 8*hi + j];
    }

    acc0 = wmma_bf16(ah0, bh0, acc0);
    acc0 = wmma_bf16(ah0, bl0, acc0);
    acc0 = wmma_bf16(al0, bh0, acc0);

    acc1 = wmma_bf16(ah0, bh1, acc1);
    acc1 = wmma_bf16(ah0, bl1, acc1);
    acc1 = wmma_bf16(al0, bh1, acc1);

    acc2 = wmma_bf16(ah1, bh0, acc2);
    acc2 = wmma_bf16(ah1, bl0, acc2);
    acc2 = wmma_bf16(al1, bh0, acc2);

    acc3 = wmma_bf16(ah1, bh1, acc3);
    acc3 = wmma_bf16(ah1, bl1, acc3);
    acc3 = wmma_bf16(al1, bh1, acc3);

    if (has_next) STORE_REGS(buf ^ 1);         // regs -> LDS (after frag loads)
    __syncthreads();
  }

  // Epilogue: C/D layout — VGPR r: lanes0-15 (M=r), lanes16-31 (M=8+r); N = lane&15.
  float* Pz = P + (size_t)z * GM * GN;
  #pragma unroll
  for (int ri = 0; ri < 2; ri++) {
    #pragma unroll
    for (int ci = 0; ci < 2; ci++) {
      int col  = nbase + colsel + ci * 16 + l16;
      float bv = bias ? bias[col] : 0.f;
      v8f c = (ri == 0) ? ((ci == 0) ? acc0 : acc1)
                        : ((ci == 0) ? acc2 : acc3);
      #pragma unroll
      for (int r = 0; r < 8; r++) {
        int row = mbase + rowsel + ri * 16 + r + hi * 8;
        Pz[(size_t)row * GN + col] = c[r] + bv;
      }
    }
  }
}

// ---------------------------------------------------------------------------
// Kernel 2: lp[b,n,h,m] = log(max(relu(pe[b,n,m,:] . Wpos[h,:] + bpos[h]), 1e-6))
// ---------------------------------------------------------------------------
__global__ __launch_bounds__(256) void pos_kernel(
    const float* __restrict__ pe, const float* __restrict__ Wpos,
    const float* __restrict__ bpos, float* __restrict__ lp)
{
  __shared__ float Wl[16 * 64];
  __shared__ float bl[16];
  const int t = threadIdx.x;
  #pragma unroll
  for (int i = 0; i < 4; i++) Wl[t * 4 + i] = Wpos[t * 4 + i];
  if (t < 16) bl[t] = bpos[t];
  __syncthreads();

  const int h = t & 15;
  const int r = blockIdx.x * 16 + (t >> 4);
  const float* prow = pe + (size_t)r * 64;
  const float* wrow = Wl + h * 64;
  float acc = bl[h];
  #pragma unroll
  for (int kk = 0; kk < 64; kk++) acc += prow[kk] * wrow[kk];
  acc = fmaxf(acc, 0.f);
  float lv = logf(fmaxf(acc, 1e-6f));
  int m = r % 36;
  int n = (r / 36) % 36;
  int b = r / 1296;
  lp[(((size_t)b * 36 + n) * 16 + h) * 36 + m] = lv;
}

// ---------------------------------------------------------------------------
// Kernel 3: per-(b,h) attention tail.
// ---------------------------------------------------------------------------
__global__ __launch_bounds__(128) void attn_kernel(
    const float* __restrict__ Q, const float* __restrict__ K,
    const float* __restrict__ V, const float* __restrict__ lp,
    const int* __restrict__ adj, const float* __restrict__ lba,
    const float* __restrict__ bout, float* __restrict__ out)
{
  const int b = blockIdx.x;
  const int h = blockIdx.y;
  __shared__ float qs[36 * 64];
  __shared__ float ks[36 * 64];
  __shared__ float vs[36 * 64];
  __shared__ float att[36 * 36];
  const int t = threadIdx.x;

  for (int idx = t; idx < 36 * 64; idx += 128) {
    int n = idx >> 6, e = idx & 63;
    size_t off = ((size_t)(b * 36 + n)) * 1024 + h * 64 + e;
    qs[idx] = Q[off];
    ks[idx] = K[off];
    vs[idx] = V[off];
  }
  __syncthreads();

  for (int idx = t; idx < 36 * 36; idx += 128) {
    int n = idx / 36, m = idx % 36;
    const float* qr = qs + n * 64;
    const float* kr = ks + m * 64;
    float s = 0.f;
    #pragma unroll
    for (int e = 0; e < 64; e++) s += qr[e] * kr[e];
    s *= 0.125f;                       // 1/sqrt(64)
    int am = adj[((size_t)(b * 36) + n) * 36 + m];
    float w = (am > 0)
                ? s + lp[(((size_t)b * 36 + n) * 16 + h) * 36 + m]
                : -9.0e15f;
    w += lba[((size_t)(b * 36) + n) * 36 + m];
    att[idx] = w;
  }
  __syncthreads();

  if (t < 36) {                        // stable softmax, one row per thread
    float* row = att + t * 36;
    float mx = row[0];
    for (int m = 1; m < 36; m++) mx = fmaxf(mx, row[m]);
    float sum = 0.f;
    for (int m = 0; m < 36; m++) { float e = expf(row[m] - mx); row[m] = e; sum += e; }
    float inv = 1.f / sum;
    for (int m = 0; m < 36; m++) row[m] *= inv;
  }
  __syncthreads();

  for (int idx = t; idx < 36 * 64; idx += 128) {
    int n = idx >> 6, e = idx & 63;
    const float* ar = att + n * 36;
    float acc = 0.f;
    #pragma unroll
    for (int m = 0; m < 36; m++) acc += ar[m] * vs[m * 64 + e];
    out[((size_t)(b * 36) + n) * 1024 + h * 64 + e] = acc + bout[h * 64 + e];
  }
}

// ---------------------------------------------------------------------------
extern "C" void kernel_launch(void* const* d_in, const int* in_sizes, int n_in,
                              void* d_out, int out_size, void* d_ws, size_t ws_size,
                              hipStream_t stream) {
  (void)in_sizes; (void)n_in; (void)out_size; (void)ws_size;
  const float* roi  = (const float*)d_in[0];
  const int*   adj  = (const int*)  d_in[1];
  const float* pe   = (const float*)d_in[2];
  const float* lba  = (const float*)d_in[3];
  const float* Wq   = (const float*)d_in[4];
  const float* bq   = (const float*)d_in[5];
  const float* Wk   = (const float*)d_in[6];
  const float* bk   = (const float*)d_in[7];
  const float* Wpos = (const float*)d_in[8];
  const float* bpos = (const float*)d_in[9];
  const float* Wout = (const float*)d_in[10];  // (16,64,1024) == (1024,1024) row-major
  const float* bout = (const float*)d_in[11];
  float* out = (float*)d_out;

  // workspace layout
  float*  P  = (float*)d_ws;                       // 3 * GM*GN f32
  float*  lp = P + (size_t)3 * GM * GN;            // LPN f32
  __bf16* Ah = (__bf16*)(lp + LPN);                // GM*GK bf16
  __bf16* Al = Ah + (size_t)GM * GK;
  __bf16* Wh = Al + (size_t)GM * GK;               // 3 * GN*GK bf16
  __bf16* Wl = Wh + (size_t)3 * GN * GK;

  const int nA4 = GM * GK / 4;      // 1179648
  const int nW4 = GN * GK / 4;      // 262144
  split_kernel<<<dim3((nA4 + 255) / 256), 256, 0, stream>>>(roi, Ah, Al, nA4);
  split_kernel<<<dim3((nW4 + 255) / 256), 256, 0, stream>>>(Wq,  Wh,                      Wl,                      nW4);
  split_kernel<<<dim3((nW4 + 255) / 256), 256, 0, stream>>>(Wk,  Wh + (size_t)GN * GK,    Wl + (size_t)GN * GK,    nW4);
  split_kernel<<<dim3((nW4 + 255) / 256), 256, 0, stream>>>(Wout,Wh + (size_t)2 * GN * GK,Wl + (size_t)2 * GN * GK,nW4);

  proj_gemm_kernel<<<dim3(GN / 64, GM / 64, 3), 128, 0, stream>>>(
      Ah, Al, Wh, Wl, bq, bk, P);
  pos_kernel<<<dim3((128 * 36 * 36) / 16), 256, 0, stream>>>(pe, Wpos, bpos, lp);
  attn_kernel<<<dim3(128, 16), 128, 0, stream>>>(
      P, P + (size_t)GM * GN, P + (size_t)2 * GM * GN, lp, adj, lba, bout, out);
}
